// RotaryMHA_48069273977317
// MI455X (gfx1250) — 16-bit and 8-bit matrix operand order repaired and checked by reading, compiled, not run on hardware
//
#include <hip/hip_runtime.h>

typedef __attribute__((ext_vector_type(16))) _Float16 v16h;
typedef __attribute__((ext_vector_type(8)))  _Float16 v8h;
typedef __attribute__((ext_vector_type(8)))  float    v8f;
typedef __attribute__((ext_vector_type(8)))  int      v8i;
typedef __attribute__((ext_vector_type(4)))  unsigned u32x4;
typedef __attribute__((ext_vector_type(8)))  int      i32x8;
typedef __attribute__((ext_vector_type(4)))  int      i32x4;

#define EPS_RMS 1.1920928955078125e-07f

__global__ __launch_bounds__(256) void bl_act_quant(
    const float* __restrict__ x, const float* __restrict__ g,
    signed char* __restrict__ xq, float* __restrict__ inv_a)
{
    const int D = 2048;
    const int row = blockIdx.x;
    const float* xr = x + (size_t)row * D;
    float xv[8];
    float ss = 0.f;
#pragma unroll
    for (int i = 0; i < 8; ++i) {
        float v = xr[threadIdx.x + 256 * i];
        xv[i] = v;
        ss += v * v;
    }
    __shared__ float red[256];
    red[threadIdx.x] = ss;
    __syncthreads();
    for (int s = 128; s > 0; s >>= 1) {
        if (threadIdx.x < s) red[threadIdx.x] += red[threadIdx.x + s];
        __syncthreads();
    }
    float ms = red[0] / (float)D;
    float rn = rsqrtf(ms + EPS_RMS);
    float amax = 0.f;
#pragma unroll
    for (int i = 0; i < 8; ++i) {
        float xn = xv[i] * rn * g[threadIdx.x + 256 * i];
        xv[i] = xn;
        amax = fmaxf(amax, fabsf(xn));
    }
    __syncthreads();
    red[threadIdx.x] = amax;
    __syncthreads();
    for (int s = 128; s > 0; s >>= 1) {
        if (threadIdx.x < s) red[threadIdx.x] = fmaxf(red[threadIdx.x], red[threadIdx.x + s]);
        __syncthreads();
    }
    float mx = fmaxf(red[0], 1e-5f);
    float a_scale = 127.f / mx;
    signed char* xo = xq + (size_t)row * D;
#pragma unroll
    for (int i = 0; i < 8; ++i) {
        float q = rintf(xv[i] * a_scale);
        q = fminf(fmaxf(q, -128.f), 127.f);
        xo[threadIdx.x + 256 * i] = (signed char)q;
    }
    if (threadIdx.x == 0) inv_a[row] = mx / 127.f;
}

__global__ __launch_bounds__(256) void bl_w_scale(
    const float* __restrict__ w, int n, float* __restrict__ wsc)
{
    float s = 0.f;
    for (int i = threadIdx.x; i < n; i += 256) s += fabsf(w[i]);
    __shared__ float red[256];
    red[threadIdx.x] = s;
    __syncthreads();
    for (int t = 128; t > 0; t >>= 1) {
        if (threadIdx.x < t) red[threadIdx.x] += red[threadIdx.x + t];
        __syncthreads();
    }
    if (threadIdx.x == 0) {
        float m = fmaxf(red[0] / (float)n, 1e-5f);
        wsc[0] = 1.f / m;
        wsc[1] = m;
    }
}

__global__ void bl_w_quant(const float* __restrict__ w, const float* __restrict__ wsc,
                           signed char* __restrict__ w8, int n)
{
    int i = blockIdx.x * blockDim.x + threadIdx.x;
    if (i < n) {
        float q = rintf(w[i] * wsc[0]);
        q = fminf(fmaxf(q, -1.f), 1.f);
        w8[i] = (signed char)q;
    }
}

__device__ __forceinline__ void tdm_stage_2d(unsigned lds_addr, const void* gptr,
                                             unsigned tile0_dw, unsigned tile1,
                                             unsigned stride_dw, int lane)
{
#if __has_builtin(__builtin_amdgcn_tensor_load_to_lds)
    (void)lane;
    unsigned long long ga = (unsigned long long)(uintptr_t)gptr;
    u32x4 g0;
    g0[0] = 1u;
    g0[1] = lds_addr;
    g0[2] = (unsigned)ga;
    g0[3] = (unsigned)((ga >> 32) & 0x1FFFFFFu) | (2u << 30);
    i32x8 g1;
    g1[0] = (int)(2u << 16);
    g1[1] = (int)((tile0_dw & 0xFFFFu) << 16);
    g1[2] = (int)(((tile0_dw >> 16) & 0xFFFFu) | ((tile1 & 0xFFFFu) << 16));
    g1[3] = (int)(((tile1 >> 16) & 0xFFFFu) | ((tile0_dw & 0xFFFFu) << 16));
    g1[4] = (int)(tile1 & 0xFFFFu);
    g1[5] = (int)stride_dw;
    g1[6] = 0;
    g1[7] = 0;
    i32x4 z = {0, 0, 0, 0};
#if defined(__clang_major__) && (__clang_major__ >= 23)
    i32x8 z8 = {};
    __builtin_amdgcn_tensor_load_to_lds(g0, g1, z, z, z8, 0);
#else
    __builtin_amdgcn_tensor_load_to_lds(g0, g1, z, z, 0);
#endif
#else
    signed char* l = (signed char*)(uintptr_t)lds_addr;
    (void)l;
    int rowdw = (int)tile0_dw;
    for (int i = lane; i < (int)(tile1 * tile0_dw); i += 32) {
        int r = i / rowdw, c = i % rowdw;
        ((__attribute__((address_space(3))) int*)(uintptr_t)lds_addr)[i] =
            *(const int*)((const char*)gptr + (size_t)r * (stride_dw * 4) + c * 4);
    }
#endif
}

__device__ __forceinline__ void tdm_wait0()
{
#if __has_builtin(__builtin_amdgcn_s_wait_tensorcnt)
    __builtin_amdgcn_s_wait_tensorcnt(0);
#else
    asm volatile("s_wait_tensorcnt 0" ::: "memory");
#endif
}

__global__ __launch_bounds__(256) void bl_gemm_i8_tdm(
    const signed char* __restrict__ A8, const signed char* __restrict__ W8,
    const float* __restrict__ inv_a, const float* __restrict__ wsc,
    float* __restrict__ out, int M, int N, int K)
{
    __shared__ __align__(16) signed char As[2][128 * 64];
    __shared__ __align__(16) signed char Bs[2][64 * 64];
    const int lane = threadIdx.x & 31;
    const int wid = threadIdx.x >> 5;
    const int nBlk = N >> 6;
    const int tm0 = (blockIdx.x / nBlk) * 128;
    const int tn0 = (blockIdx.x % nBlk) * 64;
    const int half = lane >> 4, lm = lane & 15;
    const int nch = K >> 6;
    const unsigned strideDw = (unsigned)(K >> 2);

    if (wid == 0) {
        tdm_stage_2d((unsigned)(uintptr_t)&As[0][0], A8 + (size_t)tm0 * K, 16, 128, strideDw, lane);
        tdm_stage_2d((unsigned)(uintptr_t)&Bs[0][0], W8 + (size_t)tn0 * K, 16, 64, strideDw, lane);
        tdm_wait0();
    }
    __syncthreads();

    v8i acc[4];
    acc[0] = {}; acc[1] = {}; acc[2] = {}; acc[3] = {};

    for (int c = 0; c < nch; ++c) {
        const int p = c & 1;
        if (wid == 0 && c + 1 < nch) {
            const size_t k1 = (size_t)(c + 1) << 6;
            tdm_stage_2d((unsigned)(uintptr_t)&As[p ^ 1][0], A8 + (size_t)tm0 * K + k1, 16, 128, strideDw, lane);
            tdm_stage_2d((unsigned)(uintptr_t)&Bs[p ^ 1][0], W8 + (size_t)tn0 * K + k1, 16, 64, strideDw, lane);
        }
        const signed char* arow = &As[p][(wid * 16 + lm) * 64];
        v8i a;
        {
            long long d0 = *(const long long*)(arow + half * 16);
            long long d1 = *(const long long*)(arow + 8 + half * 16);
            long long d2 = *(const long long*)(arow + 32 + half * 16);
            long long d3 = *(const long long*)(arow + 40 + half * 16);
            a[0] = (int)d0; a[1] = (int)(d0 >> 32);
            a[2] = (int)d1; a[3] = (int)(d1 >> 32);
            a[4] = (int)d2; a[5] = (int)(d2 >> 32);
            a[6] = (int)d3; a[7] = (int)(d3 >> 32);
        }
#pragma unroll
        for (int nt = 0; nt < 4; ++nt) {
            const signed char* brow = &Bs[p][(nt * 16 + lm) * 64];
            i32x4 lo = *(const i32x4*)(brow + half * 16);
            i32x4 hi = *(const i32x4*)(brow + 32 + half * 16);
            v8i b;
            b[0] = lo[0]; b[1] = lo[1]; b[2] = lo[2]; b[3] = lo[3];
            b[4] = hi[0]; b[5] = hi[1]; b[6] = hi[2]; b[7] = hi[3];
            acc[nt] = __builtin_amdgcn_wmma_i32_16x16x64_iu8(true, a, true, b, acc[nt], false, false);
        }
        if (wid == 0) tdm_wait0();
        __syncthreads();
    }

    const float invw = wsc[1];
#pragma unroll
    for (int nt = 0; nt < 4; ++nt)
#pragma unroll
        for (int r = 0; r < 8; ++r) {
            int m = tm0 + wid * 16 + half * 8 + r;
            int n = tn0 + nt * 16 + lm;
            out[(size_t)m * N + n] = (float)acc[nt][r] * inv_a[m] * invw;
        }
}

__global__ void rope_cast_f16(const float* __restrict__ src, _Float16* __restrict__ dst,
                              int T, int H)
{
    int idx = blockIdx.x * blockDim.x + threadIdx.x;
    int total = 2 * T * H * 64;
    if (idx >= total) return;
    int p = idx & 63;
    int row = idx >> 6;
    int h = row % H;
    int t = (row / H) % T;
    int b = row / (H * T);
    float inv_freq = __powf(10000.f, -(float)(2 * p) / 128.f);
    float ang = (float)t * inv_freq;
    float c = __cosf(ang), s = __sinf(ang);
    const float* sp = src + ((size_t)(b * T + t) * H + h) * 128;
    float t1 = sp[2 * p], t2 = sp[2 * p + 1];
    _Float16* dp = dst + ((size_t)(b * H + h) * T + t) * 128;
    dp[p]      = (_Float16)(t1 * c - t2 * s);
    dp[p + 64] = (_Float16)(t1 * s + t2 * c);
}

__global__ void v_cast_tr(const float* __restrict__ src, _Float16* __restrict__ dst, int T)
{
    int idx = blockIdx.x * blockDim.x + threadIdx.x;
    int total = 2 * 4 * 128 * T;
    if (idx >= total) return;
    int t = idx % T;
    int d = (idx / T) % 128;
    int kv = (idx / (T * 128)) % 4;
    int b = idx / (T * 128 * 4);
    dst[idx] = (_Float16)src[((size_t)(b * T + t) * 4 + kv) * 128 + d];
}

__global__ __launch_bounds__(256) void flash_attn_f16(
    const _Float16* __restrict__ Q,
    const _Float16* __restrict__ Kc,
    const _Float16* __restrict__ Vt,
    float* __restrict__ O,
    int T)
{
    const int H = 16, NKV = 4;
    __shared__ __align__(16) _Float16 plds[8][16 * 40];
    const int lane = threadIdx.x & 31;
    const int wid = threadIdx.x >> 5;
    const int tilesQ = T >> 4;
    const int gw = blockIdx.x * 8 + wid;
    const int q0 = (gw % tilesQ) * 16;
    const int h = (gw / tilesQ) % H;
    const int b = gw / (tilesQ * H);
    const int kv = h / (H / NKV);
    const int half = lane >> 4, lm = lane & 15;

    const _Float16* qbase = Q + ((size_t)(b * H + h) * T + q0) * 128;
    v16h qf[4];
#pragma unroll
    for (int c = 0; c < 4; ++c) {
        const _Float16* qr = qbase + (size_t)lm * 128 + c * 32 + half * 16;
        v8h lo = *(const v8h*)qr;
        v8h hi = *(const v8h*)(qr + 8);
#pragma unroll
        for (int i = 0; i < 8; ++i) { qf[c][i] = lo[i]; qf[c][i + 8] = hi[i]; }
    }
    const _Float16* kbase = Kc + (size_t)(b * NKV + kv) * T * 128;
    const _Float16* vbase = Vt + (size_t)(b * NKV + kv) * 128 * T;

    v8f o[8];
    v8f mrow, lrow;
#pragma unroll
    for (int i = 0; i < 8; ++i) o[i] = {};
#pragma unroll
    for (int r = 0; r < 8; ++r) { mrow[r] = -3.0e38f; lrow[r] = 0.f; }
    const float scale = 0.08838834764831845f;

    for (int kj = 0; kj < q0 + 16; kj += 32) {
        __builtin_prefetch(kbase + (size_t)(kj + 32 + lm) * 128, 0, 1);
        v8f s0 = {}, s1 = {};
#pragma unroll
        for (int c = 0; c < 4; ++c) {
            const _Float16* kr0 = kbase + (size_t)(kj + lm) * 128 + c * 32 + half * 16;
            const _Float16* kr1 = kbase + (size_t)(kj + 16 + lm) * 128 + c * 32 + half * 16;
            v16h b0 = *(const v16h*)kr0;
            v16h b1 = *(const v16h*)kr1;
            s0 = __builtin_amdgcn_wmma_f32_16x16x32_f16(false, qf[c], false, b0, (short)0, s0, false, false);
            s1 = __builtin_amdgcn_wmma_f32_16x16x32_f16(false, qf[c], false, b1, (short)0, s1, false, false);
        }
        v8f rm;
#pragma unroll
        for (int r = 0; r < 8; ++r) {
            int row = q0 + half * 8 + r;
            float a0 = s0[r] * scale;
            float a1 = s1[r] * scale;
            if (kj + lm > row)      a0 = -1e30f;
            if (kj + 16 + lm > row) a1 = -1e30f;
            s0[r] = a0; s1[r] = a1;
            rm[r] = fmaxf(a0, a1);
        }
#pragma unroll
        for (int msk = 1; msk <= 8; msk <<= 1)
#pragma unroll
            for (int r = 0; r < 8; ++r)
                rm[r] = fmaxf(rm[r], __shfl_xor(rm[r], msk, 32));
        v8f fac, rs;
#pragma unroll
        for (int r = 0; r < 8; ++r) {
            float mn = fmaxf(mrow[r], rm[r]);
            fac[r] = __expf(mrow[r] - mn);
            s0[r] = __expf(s0[r] - mn);
            s1[r] = __expf(s1[r] - mn);
            rs[r] = s0[r] + s1[r];
            mrow[r] = mn;
        }
#pragma unroll
        for (int msk = 1; msk <= 8; msk <<= 1)
#pragma unroll
            for (int r = 0; r < 8; ++r)
                rs[r] += __shfl_xor(rs[r], msk, 32);
#pragma unroll
        for (int r = 0; r < 8; ++r) lrow[r] = lrow[r] * fac[r] + rs[r];
#pragma unroll
        for (int t8 = 0; t8 < 8; ++t8)
#pragma unroll
            for (int r = 0; r < 8; ++r) o[t8][r] *= fac[r];

        _Float16* pl = &plds[wid][0];
        asm volatile("s_wait_dscnt 0" ::: "memory");
#pragma unroll
        for (int r = 0; r < 8; ++r) {
            pl[(half * 8 + r) * 40 + lm]      = (_Float16)s0[r];
            pl[(half * 8 + r) * 40 + 16 + lm] = (_Float16)s1[r];
        }
        asm volatile("s_wait_dscnt 0" ::: "memory");
        v16h pf;
        {
            const _Float16* pr = pl + lm * 40 + half * 16;
            v8h lo = *(const v8h*)pr;
            v8h hi = *(const v8h*)(pr + 8);
#pragma unroll
            for (int i = 0; i < 8; ++i) { pf[i] = lo[i]; pf[i + 8] = hi[i]; }
        }
#pragma unroll
        for (int t8 = 0; t8 < 8; ++t8) {
            const _Float16* vr = vbase + (size_t)(t8 * 16 + lm) * T + kj + half * 16;
            v16h bv = *(const v16h*)vr;
            o[t8] = __builtin_amdgcn_wmma_f32_16x16x32_f16(false, pf, false, bv, (short)0, o[t8], false, false);
        }
    }
#pragma unroll
    for (int r = 0; r < 8; ++r) lrow[r] = 1.f / lrow[r];
#pragma unroll
    for (int t8 = 0; t8 < 8; ++t8)
#pragma unroll
        for (int r = 0; r < 8; ++r) {
            int t = q0 + half * 8 + r;
            O[(size_t)(b * T + t) * 2048 + h * 128 + t8 * 16 + lm] = o[t8][r] * lrow[r];
        }
}

extern "C" void kernel_launch(void* const* d_in, const int* in_sizes, int n_in,
                              void* d_out, int out_size, void* d_ws, size_t ws_size,
                              hipStream_t stream)
{
    (void)in_sizes; (void)n_in; (void)out_size; (void)ws_size;
    const int B = 2, T = 2048, D = 2048, H = 16, NKV = 4;
    const int BT = B * T;
    const int KVO = NKV * 128;

    const float* x  = (const float*)d_in[0];
    const float* wq = (const float*)d_in[1];
    const float* gq = (const float*)d_in[2];
    const float* wk = (const float*)d_in[3];
    const float* gk = (const float*)d_in[4];
    const float* wv = (const float*)d_in[5];
    const float* gv = (const float*)d_in[6];
    const float* wo = (const float*)d_in[7];
    const float* go = (const float*)d_in[8];
    float* out = (float*)d_out;

    size_t off = 0;
    auto alloc = [&](size_t nbytes) -> char* {
        char* p = (char*)d_ws + off;
        off += (nbytes + 255) & ~(size_t)255;
        return p;
    };
    signed char* xq8  = (signed char*)alloc((size_t)BT * D);
    float*       inva = (float*)alloc((size_t)BT * 4);
    signed char* w8   = (signed char*)alloc((size_t)D * D);
    float*       wsc  = (float*)alloc(256);
    float*       qf   = (float*)alloc((size_t)BT * D * 4);
    float*       kf   = (float*)alloc((size_t)BT * KVO * 4);
    float*       vf   = (float*)alloc((size_t)BT * KVO * 4);
    _Float16*    q16  = (_Float16*)alloc((size_t)BT * D * 2);
    _Float16*    k16  = (_Float16*)alloc((size_t)BT * KVO * 2);
    _Float16*    vt16 = (_Float16*)alloc((size_t)BT * KVO * 2);
    float* attn = qf;

    bl_act_quant<<<BT, 256, 0, stream>>>(x, gq, xq8, inva);
    bl_w_scale<<<1, 256, 0, stream>>>(wq, D * D, wsc);
    bl_w_quant<<<(D * D + 255) / 256, 256, 0, stream>>>(wq, wsc, w8, D * D);
    bl_gemm_i8_tdm<<<(BT / 128) * (D / 64), 256, 0, stream>>>(xq8, w8, inva, wsc, qf, BT, D, D);
    bl_act_quant<<<BT, 256, 0, stream>>>(x, gk, xq8, inva);
    bl_w_scale<<<1, 256, 0, stream>>>(wk, KVO * D, wsc);
    bl_w_quant<<<(KVO * D + 255) / 256, 256, 0, stream>>>(wk, wsc, w8, KVO * D);
    bl_gemm_i8_tdm<<<(BT / 128) * (KVO / 64), 256, 0, stream>>>(xq8, w8, inva, wsc, kf, BT, KVO, D);
    bl_act_quant<<<BT, 256, 0, stream>>>(x, gv, xq8, inva);
    bl_w_scale<<<1, 256, 0, stream>>>(wv, KVO * D, wsc);
    bl_w_quant<<<(KVO * D + 255) / 256, 256, 0, stream>>>(wv, wsc, w8, KVO * D);
    bl_gemm_i8_tdm<<<(BT / 128) * (KVO / 64), 256, 0, stream>>>(xq8, w8, inva, wsc, vf, BT, KVO, D);

    int nq = B * T * H * 64;
    rope_cast_f16<<<(nq + 255) / 256, 256, 0, stream>>>(qf, q16, T, H);
    int nk = B * T * NKV * 64;
    rope_cast_f16<<<(nk + 255) / 256, 256, 0, stream>>>(kf, k16, T, NKV);
    int nv = B * NKV * 128 * T;
    v_cast_tr<<<(nv + 255) / 256, 256, 0, stream>>>(vf, vt16, T);

    flash_attn_f16<<<(B * H * (T / 16)) / 8, 256, 0, stream>>>(q16, k16, vt16, attn, T);

    bl_act_quant<<<BT, 256, 0, stream>>>(attn, go, xq8, inva);
    bl_w_scale<<<1, 256, 0, stream>>>(wo, D * D, wsc);
    bl_w_quant<<<(D * D + 255) / 256, 256, 0, stream>>>(wo, wsc, w8, D * D);
    bl_gemm_i8_tdm<<<(BT / 128) * (D / 64), 256, 0, stream>>>(xq8, w8, inva, wsc, out, BT, D, D);
}
